// FeatureInteraction_MotionRefinement_31490700214814
// MI455X (gfx1250) — matrix operand order repaired and checked by reading, not compiled, not run on hardware
//
#include <hip/hip_runtime.h>
#include <hip/hip_bf16.h>

// ---------------------------------------------------------------------------
// MI455X / gfx1250 implementation of RAFT-correlation + motion encoder +
// single-level deformable-attention encoder layer.
// All GEMM-shaped work goes through one wave32 WMMA kernel
// (v_wmma_f32_16x16x32_f16) with the B-panel staged into LDS via the CDNA5
// async global->LDS path (global_load_async_to_lds_b128 / ASYNCcnt).
// ---------------------------------------------------------------------------

typedef __attribute__((ext_vector_type(16))) _Float16 v16h;
typedef __attribute__((ext_vector_type(8)))  _Float16 v8h;
typedef __attribute__((ext_vector_type(8)))  float    v8f;

#define BB   4
#define CC   256
#define HH   48
#define WW2  48
#define HWP  (HH * WW2)     // 2304
#define TT   (BB * HWP)     // 9216

#define BROW 40             // padded LDS row stride (halves): 80B -> conflict-free

union FragU { v16h v; v8h h[2]; };

typedef __attribute__((address_space(3))) _Float16 lds_f16;

// ---------------------------------------------------------------------------
// Generic WMMA GEMM:  D[M,N] (f32 and/or f16) = A[M,K]f16 * Bt[N,K]f16^T + bias
// A row-major [M,K]; B supplied pre-transposed row-major [N,K].
// 256 threads = 8 waves; each wave computes one 16x16 tile; block covers
// 128 rows x 16 cols.  The 16x32 B-slab per K-step is staged into LDS with
// async DMA (double buffered), shared by all 8 waves.
// grid.z = batch with element strides sA/sB/sD.
// ---------------------------------------------------------------------------
__global__ void k_gemm(const _Float16* __restrict__ A, const _Float16* __restrict__ Bt,
                       const float* __restrict__ bias, float* __restrict__ D,
                       _Float16* __restrict__ D16,
                       int M, int N, int K, int ldd,
                       long long sA, long long sB, long long sD, int relu)
{
    __shared__ _Float16 smB[2][16 * BROW];   // 2 x 1.25KB double buffer

    int tid  = threadIdx.x;
    int lane = tid & 31;
    int wave = tid >> 5;
    int n0 = blockIdx.x << 4;
    int m0 = ((blockIdx.y << 3) + wave) << 4;
    int m0c = (m0 < M) ? m0 : 0;             // keep inactive waves barrier-safe
    A  += (long long)blockIdx.z * sA;
    Bt += (long long)blockIdx.z * sB;

    // A fragment (16-bit A 16x32 layout): lane<16 -> M=lane, K = {0..7,16..23};
    // lane>=16 -> M=lane-16, K = {8..15,24..31}.
    const _Float16* pa = A + (long long)(m0c + (lane & 15)) * (long long)K + ((lane >> 4) << 3);

    // Async producer: threads 0..63 move the 16x32 B-slab (16 rows x 4 b128
    // chunks) global -> LDS.  row = tid>>2, chunk = tid&3.
    int prow = tid >> 2, pchunk = tid & 3;
    const _Float16* pg = Bt + (long long)(n0 + prow) * (long long)K + pchunk * 8;
    unsigned lds0 = (unsigned)(unsigned long long)(lds_f16*)&smB[0][prow * BROW + pchunk * 8];
    unsigned lds1 = (unsigned)(unsigned long long)(lds_f16*)&smB[1][prow * BROW + pchunk * 8];

#define ISSUE_SLAB(K0, BUF)                                                     \
    do {                                                                        \
        if (tid < 64) {                                                         \
            const _Float16* g_ = pg + (K0);                                     \
            unsigned l_ = (BUF) ? lds1 : lds0;                                  \
            asm volatile("global_load_async_to_lds_b128 %0, %1, off"            \
                         :: "v"(l_), "v"(g_) : "memory");                       \
        }                                                                       \
    } while (0)

    ISSUE_SLAB(0, 0);
    asm volatile("s_wait_asynccnt 0x0" ::: "memory");
    __syncthreads();

    v8f acc = {};
    int buf = 0;
    for (int k0 = 0; k0 < K; k0 += 32) {
        if (k0 + 32 < K) ISSUE_SLAB(k0 + 32, buf ^ 1);   // overlap DMA with math
        FragU a, b;
        a.h[0] = *(const v8h*)(pa + k0);
        a.h[1] = *(const v8h*)(pa + k0 + 16);
        // B fragment (16-bit B 32x16 layout): lane<16 -> N=lane, K = {0..7,16..23};
        // lane>=16 -> N=lane-16, K = {8..15,24..31}.
        const _Float16* sb = &smB[buf][(lane & 15) * BROW + ((lane >> 4) << 3)];
        b.h[0] = *(const v8h*)(sb);
        b.h[1] = *(const v8h*)(sb + 16);
        __builtin_prefetch(pa + k0 + 32, 0, 1);          // global_prefetch_b8
        acc = __builtin_amdgcn_wmma_f32_16x16x32_f16(
                  false, a.v, false, b.v, (short)0, acc, false, false);
        asm volatile("s_wait_asynccnt 0x0" ::: "memory");
        __syncthreads();
        buf ^= 1;
    }
#undef ISSUE_SLAB

    // C/D layout: VGPR i, lanes 0-15 -> M = i, N = lane ; lanes 16-31 -> M = 8+i.
    int col = n0 + (lane & 15);
    if (m0 >= M || col >= N) return;
    float bb = bias ? bias[col] : 0.0f;
    int rbase = m0 + ((lane >> 4) << 3);
#pragma unroll
    for (int i = 0; i < 8; ++i) {
        int r = rbase + i;
        if (r >= M) break;
        float v = acc[i] + bb;
        if (relu) v = fmaxf(v, 0.0f);
        long long o = (long long)blockIdx.z * sD + (long long)r * ldd + col;
        if (D)   D[o]   = v;
        if (D16) D16[o] = (_Float16)v;
    }
}

// ---------------------------------------------------------------------------
// Helpers
// ---------------------------------------------------------------------------
__device__ __forceinline__ float bilin_zero_s(const float* __restrict__ img, int H, int W,
                                              int stride, float px, float py)
{
    float x0f = floorf(px), y0f = floorf(py);
    float wx = px - x0f, wy = py - y0f;
    int x0 = (int)x0f, y0 = (int)y0f;
    float s = 0.0f;
#pragma unroll
    for (int dy = 0; dy < 2; ++dy)
#pragma unroll
        for (int dx = 0; dx < 2; ++dx) {
            int xi = x0 + dx, yi = y0 + dy;
            if (xi >= 0 && xi < W && yi >= 0 && yi < H) {
                float wgt = (dx ? wx : 1.0f - wx) * (dy ? wy : 1.0f - wy);
                s += wgt * img[(long long)(yi * W + xi) * stride];
            }
        }
    return s;
}

// [B,C,HW] f32 -> [B,HW,C] f16 (scaled), optional f32 copy
__global__ void k_pack_fmap(const float* __restrict__ src, _Float16* __restrict__ dst16,
                            float* __restrict__ dstf, float scale, long long total)
{
    long long idx = (long long)blockIdx.x * 256 + threadIdx.x;
    if (idx >= total) return;
    int c = (int)(idx % CC);
    long long t = idx / CC;
    int b = (int)(t / HWP), p = (int)(t % HWP);
    float v = src[((long long)b * CC + c) * HWP + p] * scale;
    dst16[idx] = (_Float16)v;
    if (dstf) dstf[idx] = v;
}

// OIHW f32 -> padded [Npad, Kpad] f16, col = (ky*KW+kx)*I + c
__global__ void k_pack_wconv(const float* __restrict__ w, _Float16* __restrict__ dst,
                             int O, int I, int KH, int KW, int Npad, int Kpad, long long total)
{
    long long idx = (long long)blockIdx.x * 256 + threadIdx.x;
    if (idx >= total) return;
    int col = (int)(idx % Kpad);
    int o   = (int)(idx / Kpad);
    float v = 0.0f;
    int tap = col / I, c = col % I;
    if (o < O && tap < KH * KW) {
        int ky = tap / KW, kx = tap % KW;
        v = w[(((long long)o * I + c) * KH + ky) * KW + kx];
    }
    dst[idx] = (_Float16)v;
}

__global__ void k_pack_lin(const float* __restrict__ w, _Float16* __restrict__ dst, long long total)
{
    long long idx = (long long)blockIdx.x * 256 + threadIdx.x;
    if (idx >= total) return;
    dst[idx] = (_Float16)w[idx];
}

// correlation lookup: 81 bilinear taps per token -> im2col-ready [T, 96] f16
__global__ void k_corr_lookup(const float* __restrict__ corr, const float* __restrict__ flow,
                              _Float16* __restrict__ col81, long long total)
{
    long long idx = (long long)blockIdx.x * 256 + threadIdx.x;
    if (idx >= total) return;
    int tap = (int)(idx % 96);
    long long t = idx / 96;
    if (tap >= 81) { col81[idx] = (_Float16)0.0f; return; }
    int b = (int)(t / HWP), p = (int)(t % HWP);
    int y = p / WW2, x = p % WW2;
    float cx = (float)x + flow[((long long)b * 2 + 0) * HWP + p];
    float cy = (float)y + flow[((long long)b * 2 + 1) * HWP + p];
    // RAFT quirk: px offset = d[tap/9], py offset = d[tap%9]
    float px = cx + (float)(tap / 9 - 4);
    float py = cy + (float)(tap % 9 - 4);
    const float* img = corr + ((long long)b * HWP + p) * HWP;
    col81[idx] = (_Float16)bilin_zero_s(img, HH, WW2, 1, px, py);
}

// 3x3 im2col from token-major f16 activations, zero-pad borders
__global__ void k_im2col3(const _Float16* __restrict__ src, _Float16* __restrict__ dst,
                          int Cin, long long total)
{
    long long idx = (long long)blockIdx.x * 256 + threadIdx.x;
    if (idx >= total) return;
    int K = 9 * Cin;
    int col = (int)(idx % K);
    long long t = idx / K;
    int b = (int)(t / HWP), p = (int)(t % HWP);
    int y = p / WW2, x = p % WW2;
    int tap = col / Cin, c = col % Cin;
    int yy = y + tap / 3 - 1, xx = x + tap % 3 - 1;
    _Float16 v = (_Float16)0.0f;
    if (yy >= 0 && yy < HH && xx >= 0 && xx < WW2)
        v = src[((long long)b * HWP + yy * WW2 + xx) * Cin + c];
    dst[idx] = v;
}

// 7x7 im2col of flow [B,2,H,W] f32 -> [T, 128] f16 (K=98 zero-padded to 128)
__global__ void k_im2col_flow(const float* __restrict__ flow, _Float16* __restrict__ dst,
                              long long total)
{
    long long idx = (long long)blockIdx.x * 256 + threadIdx.x;
    if (idx >= total) return;
    int col = (int)(idx % 128);
    long long t = idx / 128;
    float v = 0.0f;
    if (col < 98) {
        int b = (int)(t / HWP), p = (int)(t % HWP);
        int y = p / WW2, x = p % WW2;
        int tap = col / 2, c = col % 2;
        int yy = y + tap / 7 - 3, xx = x + tap % 7 - 3;
        if (yy >= 0 && yy < HH && xx >= 0 && xx < WW2)
            v = flow[(((long long)b * 2 + c) * HH + yy) * WW2 + xx];
    }
    dst[idx] = (_Float16)v;
}

__global__ void k_concat2(const _Float16* __restrict__ a, const _Float16* __restrict__ b2,
                          _Float16* __restrict__ dst, int Ca, int Cb, long long total)
{
    long long idx = (long long)blockIdx.x * 256 + threadIdx.x;
    if (idx >= total) return;
    int C = Ca + Cb;
    int c = (int)(idx % C);
    long long t = idx / C;
    dst[idx] = (c < Ca) ? a[t * Ca + c] : b2[t * Cb + (c - Ca)];
}

// append flow channels (cols 126,127) into mf16 [T,128]
__global__ void k_append_flow(const float* __restrict__ flow, _Float16* __restrict__ mf16,
                              long long total)
{
    long long idx = (long long)blockIdx.x * 256 + threadIdx.x;
    if (idx >= total) return;
    int ch = (int)(idx % 2);
    long long t = idx / 2;
    int b = (int)(t / HWP), p = (int)(t % HWP);
    mf16[t * 128 + 126 + ch] = (_Float16)flow[((long long)b * 2 + ch) * HWP + p];
}

// q = src + sine positional embedding (normalize=True), f16
__global__ void k_pos_q(const float* __restrict__ srcF, _Float16* __restrict__ q16,
                        long long total)
{
    long long idx = (long long)blockIdx.x * 256 + threadIdx.x;
    if (idx >= total) return;
    int c = (int)(idx % CC);
    long long t = idx / CC;
    int p = (int)(t % HWP);
    int y = p / WW2, x = p % WW2;
    const float twopi = 6.283185307179586f;
    float coord; int j;
    if (c < 128) { j = c;       coord = ((float)y + 0.5f) / ((float)HH + 1e-6f) * twopi; }
    else         { j = c - 128; coord = ((float)x + 0.5f) / ((float)WW2 + 1e-6f) * twopi; }
    float e = powf(10000.0f, (float)(j & ~1) / 128.0f);
    float v = coord / e;
    float pos = (j & 1) ? cosf(v) : sinf(v);
    q16[idx] = (_Float16)(srcF[idx] + pos);
}

// softmax over groups of 4 (NH*NP attention weights)
__global__ void k_softmax4(const float* __restrict__ x, float* __restrict__ y, long long total)
{
    long long idx = (long long)blockIdx.x * 256 + threadIdx.x;
    if (idx >= total) return;
    const float* p = x + idx * 4;
    float m = fmaxf(fmaxf(p[0], p[1]), fmaxf(p[2], p[3]));
    float e0 = __expf(p[0] - m), e1 = __expf(p[1] - m);
    float e2 = __expf(p[2] - m), e3 = __expf(p[3] - m);
    float s = 1.0f / (e0 + e1 + e2 + e3);
    float* q = y + idx * 4;
    q[0] = e0 * s; q[1] = e1 * s; q[2] = e2 * s; q[3] = e3 * s;
}

// deformable attention sampling: per (token, channel) gather 4 points x 4 taps
__global__ void k_deform(const float* __restrict__ offb, const float* __restrict__ awn,
                         const float* __restrict__ valb, _Float16* __restrict__ attn16,
                         long long total)
{
    long long idx = (long long)blockIdx.x * 256 + threadIdx.x;
    if (idx >= total) return;
    int cg = (int)(idx % CC);
    long long t = idx / CC;
    int h = cg >> 5;
    int b = (int)(t / HWP), p = (int)(t % HWP);
    int y = p / WW2, x = p % WW2;
    const float* base = valb + (long long)b * HWP * CC + cg;
    float acc = 0.0f;
#pragma unroll
    for (int pt = 0; pt < 4; ++pt) {
        float ox = offb[t * 64 + h * 8 + pt * 2 + 0];
        float oy = offb[t * 64 + h * 8 + pt * 2 + 1];
        float w4 = awn[t * 32 + h * 4 + pt];
        // normalized-loc algebra collapses to: px = x + off_x, py = y + off_y
        acc += w4 * bilin_zero_s(base, HH, WW2, CC, (float)x + ox, (float)y + oy);
    }
    attn16[idx] = (_Float16)acc;
}

// y = LayerNorm(x + r) over C=256; one wave (32 lanes) per token
__global__ void k_add_ln(const float* __restrict__ x, const float* __restrict__ r,
                         const float* __restrict__ g, const float* __restrict__ bta,
                         float* __restrict__ y, _Float16* __restrict__ y16)
{
    int token = blockIdx.x * 8 + (threadIdx.x >> 5);
    int lane = threadIdx.x & 31;
    const float* xr = x + (long long)token * CC;
    const float* rr = r + (long long)token * CC;
    float v[8], s = 0.0f, s2 = 0.0f;
#pragma unroll
    for (int i = 0; i < 8; ++i) {
        int c = lane + 32 * i;
        float vv = xr[c] + rr[c];
        v[i] = vv; s += vv; s2 += vv * vv;
    }
#pragma unroll
    for (int m = 16; m >= 1; m >>= 1) {
        s  += __shfl_xor(s,  m, 32);
        s2 += __shfl_xor(s2, m, 32);
    }
    float mean = s * (1.0f / CC);
    float var  = s2 * (1.0f / CC) - mean * mean;
    float inv  = rsqrtf(var + 1e-5f);
#pragma unroll
    for (int i = 0; i < 8; ++i) {
        int c = lane + 32 * i;
        float o = (v[i] - mean) * inv * g[c] + bta[c];
        y[(long long)token * CC + c] = o;
        if (y16) y16[(long long)token * CC + c] = (_Float16)o;
    }
}

// delta_flow [T,2] token-major -> d_out tail as [B,2,H,W]
__global__ void k_flow_out(const float* __restrict__ df, float* __restrict__ out2, long long total)
{
    long long idx = (long long)blockIdx.x * 256 + threadIdx.x;
    if (idx >= total) return;
    int ch = (int)(idx % 2);
    long long t = idx / 2;
    int b = (int)(t / HWP), p = (int)(t % HWP);
    out2[((long long)b * 2 + ch) * HWP + p] = df[t * 2 + ch];
}

// ---------------------------------------------------------------------------
// Host orchestration
// ---------------------------------------------------------------------------
static inline dim3 eg(long long n) { return dim3((unsigned)((n + 255) / 256)); }

extern "C" void kernel_launch(void* const* d_in, const int* in_sizes, int n_in,
                              void* d_out, int out_size, void* d_ws, size_t ws_size,
                              hipStream_t stream)
{
    (void)in_sizes; (void)n_in; (void)out_size; (void)ws_size;
    const float* fmap1 = (const float*)d_in[0];
    const float* fmap2 = (const float*)d_in[1];
    const float* flow  = (const float*)d_in[2];
    const float* wc1   = (const float*)d_in[3];  const float* bc1  = (const float*)d_in[4];
    const float* wc2   = (const float*)d_in[5];  const float* bc2  = (const float*)d_in[6];
    const float* wf1   = (const float*)d_in[7];  const float* bf1  = (const float*)d_in[8];
    const float* wf2   = (const float*)d_in[9];  const float* bf2  = (const float*)d_in[10];
    const float* wcf   = (const float*)d_in[11]; const float* bcf  = (const float*)d_in[12];
    const float* wfh1  = (const float*)d_in[13]; const float* bfh1 = (const float*)d_in[14];
    const float* wfh2  = (const float*)d_in[15]; const float* bfh2 = (const float*)d_in[16];
    const float* w_off = (const float*)d_in[17]; const float* b_off= (const float*)d_in[18];
    const float* w_aw  = (const float*)d_in[19]; const float* b_aw = (const float*)d_in[20];
    const float* w_val = (const float*)d_in[21]; const float* b_val= (const float*)d_in[22];
    const float* w_out = (const float*)d_in[23]; const float* b_out= (const float*)d_in[24];
    const float* ln1g  = (const float*)d_in[25]; const float* ln1b = (const float*)d_in[26];
    const float* w_ff1 = (const float*)d_in[27]; const float* b_ff1= (const float*)d_in[28];
    const float* w_ff2 = (const float*)d_in[29]; const float* b_ff2= (const float*)d_in[30];
    const float* ln2g  = (const float*)d_in[31]; const float* ln2b = (const float*)d_in[32];
    float* out = (float*)d_out;

    size_t off = 0;
    char* base = (char*)d_ws;
    auto alloc = [&](size_t bytes) -> void* {
        void* p = base + off;
        off = (off + bytes + 255) & ~(size_t)255;
        return p;
    };
    const long long T = TT, HW = HWP;

    _Float16* f1t    = (_Float16*)alloc((size_t)T * CC * 2);
    _Float16* f2t    = (_Float16*)alloc((size_t)T * CC * 2);
    float*    srcF   = (float*)   alloc((size_t)T * CC * 4);
    _Float16* src16  = (_Float16*)alloc((size_t)T * CC * 2);
    float*    corr   = (float*)   alloc((size_t)BB * HW * HW * 4);
    _Float16* col81  = (_Float16*)alloc((size_t)T * 96 * 2);
    _Float16* colbuf = (_Float16*)alloc((size_t)T * 2304 * 2);
    _Float16* colflw = (_Float16*)alloc((size_t)T * 128 * 2);
    _Float16* cor1   = (_Float16*)alloc((size_t)T * 256 * 2);
    _Float16* cor2   = (_Float16*)alloc((size_t)T * 192 * 2);
    _Float16* flo1   = (_Float16*)alloc((size_t)T * 128 * 2);
    _Float16* flo2   = (_Float16*)alloc((size_t)T * 64 * 2);
    _Float16* cat16  = (_Float16*)alloc((size_t)T * 256 * 2);
    _Float16* mf16   = (_Float16*)alloc((size_t)T * 128 * 2);
    _Float16* fh16   = (_Float16*)alloc((size_t)T * 256 * 2);
    float*    dflow  = (float*)   alloc((size_t)T * 2 * 4);
    _Float16* q16    = (_Float16*)alloc((size_t)T * 256 * 2);
    float*    offF   = (float*)   alloc((size_t)T * 64 * 4);
    float*    awF    = (float*)   alloc((size_t)T * 32 * 4);
    float*    awN    = (float*)   alloc((size_t)T * 32 * 4);
    float*    valF   = (float*)   alloc((size_t)T * 256 * 4);
    _Float16* attn16 = (_Float16*)alloc((size_t)T * 256 * 2);
    float*    aoutF  = (float*)   alloc((size_t)T * 256 * 4);
    float*    src2F  = (float*)   alloc((size_t)T * 256 * 4);
    _Float16* src216 = (_Float16*)alloc((size_t)T * 256 * 2);
    _Float16* ffh16  = (_Float16*)alloc((size_t)T * 1024 * 2);
    float*    ff2F   = (float*)   alloc((size_t)T * 256 * 4);
    _Float16* wc1p   = (_Float16*)alloc((size_t)256 * 96 * 2);
    _Float16* wc2p   = (_Float16*)alloc((size_t)192 * 2304 * 2);
    _Float16* wf1p   = (_Float16*)alloc((size_t)128 * 128 * 2);
    _Float16* wf2p   = (_Float16*)alloc((size_t)64 * 1152 * 2);
    _Float16* wcfp   = (_Float16*)alloc((size_t)128 * 2304 * 2);
    _Float16* wfh1p  = (_Float16*)alloc((size_t)256 * 1152 * 2);
    _Float16* wfh2p  = (_Float16*)alloc((size_t)16 * 2304 * 2);
    _Float16* woffp  = (_Float16*)alloc((size_t)64 * 256 * 2);
    _Float16* wawp   = (_Float16*)alloc((size_t)32 * 256 * 2);
    _Float16* wvalp  = (_Float16*)alloc((size_t)256 * 256 * 2);
    _Float16* woutp  = (_Float16*)alloc((size_t)256 * 256 * 2);
    _Float16* wff1p  = (_Float16*)alloc((size_t)1024 * 256 * 2);
    _Float16* wff2p  = (_Float16*)alloc((size_t)256 * 1024 * 2);

    // ---- pack features / weights (f32 -> f16, WMMA-ready layouts) ----
    long long nTC = T * CC;
    k_pack_fmap<<<eg(nTC), 256, 0, stream>>>(fmap1, f1t, nullptr, 0.0625f, nTC); // 1/sqrt(256)
    k_pack_fmap<<<eg(nTC), 256, 0, stream>>>(fmap2, f2t, nullptr, 1.0f, nTC);
    k_pack_fmap<<<eg(nTC), 256, 0, stream>>>(fmap1, src16, srcF, 1.0f, nTC);
    k_pack_wconv<<<eg(256*96),  256, 0, stream>>>(wc1,  wc1p, 256, 81, 1, 1, 256, 96,   256*96);
    k_pack_wconv<<<eg(192*2304),256, 0, stream>>>(wc2,  wc2p, 192, 256,3, 3, 192, 2304, 192*2304);
    k_pack_wconv<<<eg(128*128), 256, 0, stream>>>(wf1,  wf1p, 128, 2,  7, 7, 128, 128,  128*128);
    k_pack_wconv<<<eg(64*1152), 256, 0, stream>>>(wf2,  wf2p, 64,  128,3, 3, 64,  1152, 64*1152);
    k_pack_wconv<<<eg(128*2304),256, 0, stream>>>(wcf,  wcfp, 126, 256,3, 3, 128, 2304, 128*2304);
    k_pack_wconv<<<eg(256*1152),256, 0, stream>>>(wfh1, wfh1p,256, 128,3, 3, 256, 1152, 256*1152);
    k_pack_wconv<<<eg(16*2304), 256, 0, stream>>>(wfh2, wfh2p,2,   256,3, 3, 16,  2304, 16*2304);
    k_pack_lin<<<eg(64*256),  256, 0, stream>>>(w_off, woffp, 64*256);
    k_pack_lin<<<eg(32*256),  256, 0, stream>>>(w_aw,  wawp,  32*256);
    k_pack_lin<<<eg(256*256), 256, 0, stream>>>(w_val, wvalp, 256*256);
    k_pack_lin<<<eg(256*256), 256, 0, stream>>>(w_out, woutp, 256*256);
    k_pack_lin<<<eg(1024*256),256, 0, stream>>>(w_ff1, wff1p, 1024*256);
    k_pack_lin<<<eg(256*1024),256, 0, stream>>>(w_ff2, wff2p, 256*1024);

    // ---- all-pairs correlation: per-batch [2304,256] x [256,2304] WMMA GEMM ----
    k_gemm<<<dim3(HWP/16, HWP/128, BB), 256, 0, stream>>>(
        f1t, f2t, nullptr, corr, nullptr, HWP, HWP, 256, HWP,
        (long long)HW*256, (long long)HW*256, (long long)HW*HW, 0);

    // ---- correlation lookup (corr volume lives in 192MB L2) ----
    k_corr_lookup<<<eg(T*96), 256, 0, stream>>>(corr, flow, col81, T*96);

    // ---- motion encoder (im2col + WMMA GEMM chain) ----
    k_gemm<<<dim3(16, 72, 1), 256, 0, stream>>>(col81, wc1p, bc1, nullptr, cor1,
        TT, 256, 96, 256, 0, 0, 0, 1);
    k_im2col3<<<eg(T*2304), 256, 0, stream>>>(cor1, colbuf, 256, T*2304);
    k_gemm<<<dim3(12, 72, 1), 256, 0, stream>>>(colbuf, wc2p, bc2, nullptr, cor2,
        TT, 192, 2304, 192, 0, 0, 0, 1);
    k_im2col_flow<<<eg(T*128), 256, 0, stream>>>(flow, colflw, T*128);
    k_gemm<<<dim3(8, 72, 1), 256, 0, stream>>>(colflw, wf1p, bf1, nullptr, flo1,
        TT, 128, 128, 128, 0, 0, 0, 1);
    k_im2col3<<<eg(T*1152), 256, 0, stream>>>(flo1, colbuf, 128, T*1152);
    k_gemm<<<dim3(4, 72, 1), 256, 0, stream>>>(colbuf, wf2p, bf2, nullptr, flo2,
        TT, 64, 1152, 64, 0, 0, 0, 1);
    k_concat2<<<eg(T*256), 256, 0, stream>>>(cor2, flo2, cat16, 192, 64, T*256);
    k_im2col3<<<eg(T*2304), 256, 0, stream>>>(cat16, colbuf, 256, T*2304);
    k_gemm<<<dim3(8, 72, 1), 256, 0, stream>>>(colbuf, wcfp, bcf, nullptr, mf16,
        TT, 126, 2304, 128, 0, 0, 0, 1);
    k_append_flow<<<eg(T*2), 256, 0, stream>>>(flow, mf16, T*2);
    k_im2col3<<<eg(T*1152), 256, 0, stream>>>(mf16, colbuf, 128, T*1152);
    k_gemm<<<dim3(16, 72, 1), 256, 0, stream>>>(colbuf, wfh1p, bfh1, nullptr, fh16,
        TT, 256, 1152, 256, 0, 0, 0, 1);
    k_im2col3<<<eg(T*2304), 256, 0, stream>>>(fh16, colbuf, 256, T*2304);
    k_gemm<<<dim3(1, 72, 1), 256, 0, stream>>>(colbuf, wfh2p, bfh2, dflow, nullptr,
        TT, 2, 2304, 2, 0, 0, 0, 0);
    k_flow_out<<<eg(T*2), 256, 0, stream>>>(dflow, out + (long long)T*256, T*2);

    // ---- deformable attention encoder layer ----
    k_pos_q<<<eg(T*256), 256, 0, stream>>>(srcF, q16, T*256);
    k_gemm<<<dim3(4, 72, 1), 256, 0, stream>>>(q16, woffp, b_off, offF, nullptr,
        TT, 64, 256, 64, 0, 0, 0, 0);
    k_gemm<<<dim3(2, 72, 1), 256, 0, stream>>>(q16, wawp, b_aw, awF, nullptr,
        TT, 32, 256, 32, 0, 0, 0, 0);
    k_softmax4<<<eg(T*8), 256, 0, stream>>>(awF, awN, T*8);
    k_gemm<<<dim3(16, 72, 1), 256, 0, stream>>>(src16, wvalp, b_val, valF, nullptr,
        TT, 256, 256, 256, 0, 0, 0, 0);
    k_deform<<<eg(T*256), 256, 0, stream>>>(offF, awN, valF, attn16, T*256);
    k_gemm<<<dim3(16, 72, 1), 256, 0, stream>>>(attn16, woutp, b_out, aoutF, nullptr,
        TT, 256, 256, 256, 0, 0, 0, 0);
    k_add_ln<<<dim3(TT/8), 256, 0, stream>>>(aoutF, srcF, ln1g, ln1b, src2F, src216);
    k_gemm<<<dim3(64, 72, 1), 256, 0, stream>>>(src216, wff1p, b_ff1, nullptr, ffh16,
        TT, 1024, 256, 1024, 0, 0, 0, 1);
    k_gemm<<<dim3(16, 72, 1), 256, 0, stream>>>(ffh16, wff2p, b_ff2, ff2F, nullptr,
        TT, 256, 1024, 256, 0, 0, 0, 0);
    k_add_ln<<<dim3(TT/8), 256, 0, stream>>>(ff2F, src2F, ln2g, ln2b, out, nullptr);
}
